// MambaRegressor_66778151518288
// MI455X (gfx1250) — matrix operand order repaired and checked by reading, not compiled, not run on hardware
//
#include <hip/hip_runtime.h>

// ---------------- dims (match reference) ----------------
#define BATCH   32
#define CTX     8192
#define NPATCH  512
#define PSIZE   16
#define DMODEL  256
#define DSTATE  16
#define DCONV   4
#define DINNER  512
#define DTRANK  16
#define XZW     (2 * DINNER)          // 1024
#define XDW     (DTRANK + 2 * DSTATE) // 48
#define NROWS   (BATCH * NPATCH)      // 16384
#define EPSF    1e-5f

// ---------------- WMMA vector types (gfx1250) ----------------
typedef __attribute__((ext_vector_type(8)))  __bf16 v8bf;
typedef __attribute__((ext_vector_type(16))) __bf16 v16bf;
typedef __attribute__((ext_vector_type(8)))  float  v8f;
typedef __attribute__((ext_vector_type(4)))  unsigned int v4u;
typedef __attribute__((ext_vector_type(4)))  int v4i;

#define AS1 __attribute__((address_space(1)))
#define AS3 __attribute__((address_space(3)))

// ---- async global->LDS copy (CDNA5 ASYNCcnt path), with sync fallback ----
// probe-learned signature: (v4i AS1* gsrc, v4i AS3* ldst, Ii offset, Ii cpol)
#if __has_builtin(__builtin_amdgcn_global_load_async_to_lds_b128)
#define ASYNC_CP_B128(g, l)                                                   \
    __builtin_amdgcn_global_load_async_to_lds_b128(                           \
        (AS1 v4i*)(g), (AS3 v4i*)(l), 0, 0)
#if __has_builtin(__builtin_amdgcn_s_wait_asynccnt)
#define ASYNC_WAIT() __builtin_amdgcn_s_wait_asynccnt(0)
#else
#define ASYNC_WAIT() asm volatile("s_wait_asynccnt 0" ::: "memory")
#endif
#else
#define ASYNC_CP_B128(g, l) (*(v4u*)(l) = *(const v4u*)(g))
#define ASYNC_WAIT()        ((void)0)
#endif

__device__ __forceinline__ unsigned short f2bf(float f) {
    union { float f; unsigned u; } v; v.f = f;
    unsigned r = v.u + 0x7FFFu + ((v.u >> 16) & 1u);   // round-to-nearest-even
    return (unsigned short)(r >> 16);
}

__device__ __forceinline__ float silu(float x) {
    return x / (1.f + __expf(-x));
}

// ---------------- generic f32 -> bf16 convert ----------------
__global__ void f32_to_bf16_kernel(const float* __restrict__ src,
                                   unsigned short* __restrict__ dst, int n) {
    int i = blockIdx.x * blockDim.x + threadIdx.x;
    if (i < n) dst[i] = f2bf(src[i]);
}

// ---------------- patch embed: LN -> Linear(16,256) -> LN -> +posemb ----------------
__global__ void patch_embed_kernel(const float* __restrict__ x,
                                   const float* __restrict__ ln1_g, const float* __restrict__ ln1_b,
                                   const float* __restrict__ W_emb, const float* __restrict__ b_emb,
                                   const float* __restrict__ ln2_g, const float* __restrict__ ln2_b,
                                   unsigned short* __restrict__ Pbf) {
    int row = blockIdx.x;
    int n   = row & (NPATCH - 1);
    int d   = threadIdx.x;
    __shared__ float xv[PSIZE];
    __shared__ float red[DMODEL];
    if (d < PSIZE) xv[d] = x[(size_t)row * PSIZE + d];
    __syncthreads();

    float mu = 0.f;
    #pragma unroll
    for (int j = 0; j < PSIZE; ++j) mu += xv[j];
    mu *= (1.f / PSIZE);
    float var = 0.f;
    #pragma unroll
    for (int j = 0; j < PSIZE; ++j) { float c = xv[j] - mu; var += c * c; }
    var *= (1.f / PSIZE);
    float rs = rsqrtf(var + EPSF);

    float t = b_emb[d];
    #pragma unroll
    for (int j = 0; j < PSIZE; ++j) {
        float xn = (xv[j] - mu) * rs * ln1_g[j] + ln1_b[j];
        t += xn * W_emb[j * DMODEL + d];
    }

    red[d] = t; __syncthreads();
    for (int s = DMODEL / 2; s > 0; s >>= 1) { if (d < s) red[d] += red[d + s]; __syncthreads(); }
    float mu2 = red[0] * (1.f / DMODEL);
    __syncthreads();
    float c = t - mu2;
    red[d] = c * c; __syncthreads();
    for (int s = DMODEL / 2; s > 0; s >>= 1) { if (d < s) red[d] += red[d + s]; __syncthreads(); }
    float var2 = red[0] * (1.f / DMODEL);
    float val = c * rsqrtf(var2 + EPSF) * ln2_g[d] + ln2_b[d];

    int i = (d < DMODEL / 2) ? d : d - DMODEL / 2;
    float omega = __expf(-(float)i * (9.210340371976184f / (float)(DMODEL / 2 - 1)));
    float ang = (float)n * omega;
    val += (d < DMODEL / 2) ? __sinf(ang) : __cosf(ang);

    Pbf[(size_t)row * DMODEL + d] = f2bf(val);
}

// ---------------- blocked bf16 WMMA GEMM: C[M,N] = A[M,K] * W[N,K]^T ----------------
// Block = 128 threads (4 waves) -> 128 x (NTILES*16) output tile.
// Each wave: 2 M-tiles x NTILES N-tiles. 64x32 B-tile double-buffered in LDS via
// async global->LDS copies, shared by all 4 waves. NTILES/GUARD are compile-time,
// so the K-loop is branch-free straight-line WMMA code (EXEC all-1s throughout).
// A/B fragments: contiguous k order (k = khalf*16 + s); C/D layout per CDNA5 ISA 7.12.2 (wave32).
template <int NTILES, bool GUARD>
__global__ void gemm_bf16_wmma(const unsigned short* __restrict__ A,
                               const unsigned short* __restrict__ W,
                               float* __restrict__ C,
                               int M, int N, int K) {
    __shared__ alignas(64) unsigned short Bs[2][64 * 32];   // 4KB per buffer

    const int tid   = threadIdx.x;
    const int wave  = tid >> 5;
    const int lane  = tid & 31;
    const int khalf = lane >> 4;         // 0: K 0-15   1: K 16-31
    const int rc    = lane & 15;         // A row / B column within 16-tile
    const int n0    = blockIdx.y * 64;
    const int m0    = blockIdx.x * 128 + wave * 32;

    // A fragment pointers (2 M-tiles per wave)
    const unsigned short* pa0 = A + (size_t)(m0 + rc) * K + khalf * 16;
    const unsigned short* pa1 = pa0 + (size_t)16 * K;

    // B staging: thread copies 2 x 16B chunks per K-step (rows srow, srow+32)
    const int srow = tid >> 2;           // 0..31
    const int scol = (tid & 3) * 8;      // halfs
    const unsigned short* wsrc0 = W + (size_t)(n0 + srow) * K + scol;
    const unsigned short* wsrc1 = W + (size_t)(n0 + srow + 32) * K + scol;
    const bool g0 = !GUARD || (n0 + srow)      < N;
    const bool g1 = !GUARD || (n0 + srow + 32) < N;

#define STAGE_B(buf, kk) do {                                                 \
        if (g0) ASYNC_CP_B128(wsrc0 + (kk), &Bs[buf][srow * 32 + scol]);       \
        if (g1) ASYNC_CP_B128(wsrc1 + (kk), &Bs[buf][(srow + 32) * 32 + scol]);\
    } while (0)

    v8f zero = {};
    v8f acc[2][NTILES];
    #pragma unroll
    for (int mt = 0; mt < 2; ++mt)
        #pragma unroll
        for (int nt = 0; nt < NTILES; ++nt) acc[mt][nt] = zero;

    STAGE_B(0, 0);

    for (int k0 = 0; k0 < K; k0 += 32) {
        const int nb = (k0 >> 5) & 1;
        ASYNC_WAIT();
        __syncthreads();
        if (k0 + 32 < K) {
            STAGE_B(nb ^ 1, k0 + 32);
            __builtin_prefetch(pa0 + k0 + 32, 0, 1);   // global_prefetch_b8
            __builtin_prefetch(pa1 + k0 + 32, 0, 1);
        }

        // A fragments (contiguous layout: two 16B chunks, K = khalf*16+[0..7] / +8)
        v8bf a0lo = *(const v8bf*)(pa0 + k0);
        v8bf a0hi = *(const v8bf*)(pa0 + k0 + 8);
        v8bf a1lo = *(const v8bf*)(pa1 + k0);
        v8bf a1hi = *(const v8bf*)(pa1 + k0 + 8);
        v16bf a0 = __builtin_shufflevector(a0lo, a0hi,
                       0, 1, 2, 3, 4, 5, 6, 7, 8, 9, 10, 11, 12, 13, 14, 15);
        v16bf a1 = __builtin_shufflevector(a1lo, a1hi,
                       0, 1, 2, 3, 4, 5, 6, 7, 8, 9, 10, 11, 12, 13, 14, 15);

        #pragma unroll
        for (int nt = 0; nt < NTILES; ++nt) {
            v16bf b = *(const v16bf*)&Bs[nb][(nt * 16 + rc) * 32 + khalf * 16];
            acc[0][nt] = __builtin_amdgcn_wmma_f32_16x16x32_bf16(
                             false, a0, false, b, (short)0, acc[0][nt], false, false);
            acc[1][nt] = __builtin_amdgcn_wmma_f32_16x16x32_bf16(
                             false, a1, false, b, (short)0, acc[1][nt], false, false);
        }
    }
#undef STAGE_B

    // C/D layout: VGPR v -> row (tile_m + khalf*8 + v), col (tile_n + rc)
    #pragma unroll
    for (int mt = 0; mt < 2; ++mt) {
        #pragma unroll
        for (int nt = 0; nt < NTILES; ++nt) {
            float* pc = C + (size_t)(m0 + mt * 16 + khalf * 8) * N + n0 + nt * 16 + rc;
            #pragma unroll
            for (int v = 0; v < 8; ++v) pc[(size_t)v * N] = acc[mt][nt][v];
        }
    }
}

// ---------------- causal depthwise conv (DCONV=4) + bias + SiLU ----------------
__global__ void conv_silu_kernel(const float* __restrict__ xz,
                                 const float* __restrict__ conv_w,
                                 const float* __restrict__ conv_b,
                                 float* __restrict__ Uf,
                                 unsigned short* __restrict__ Ubf) {
    int row = blockIdx.x;            // b*NPATCH + l
    int l   = row & (NPATCH - 1);
    int d   = threadIdx.x;
    float acc = conv_b[d];
    #pragma unroll
    for (int j = 0; j < DCONV; ++j) {
        int ls = l - (DCONV - 1) + j;
        if (ls >= 0)
            acc += conv_w[d * DCONV + j] *
                   xz[(size_t)(row - (DCONV - 1) + j) * XZW + d];
    }
    float s = silu(acc);
    Uf [(size_t)row * DINNER + d] = s;
    Ubf[(size_t)row * DINNER + d] = f2bf(s);
}

// ---------------- dt = softplus(dt_in @ dt_proj_w^T + b) ----------------
__global__ void dt_kernel(const float* __restrict__ XD,
                          const float* __restrict__ dt_w,
                          const float* __restrict__ dt_b,
                          float* __restrict__ DT) {
    int row = blockIdx.x;
    int tid = threadIdx.x;           // 256
    __shared__ float xr[DTRANK];
    if (tid < DTRANK) xr[tid] = XD[(size_t)row * XDW + tid];
    __syncthreads();
    for (int d = tid; d < DINNER; d += blockDim.x) {
        float a = dt_b[d];
        #pragma unroll
        for (int r = 0; r < DTRANK; ++r) a += xr[r] * dt_w[d * DTRANK + r];
        DT[(size_t)row * DINNER + d] = (a > 20.f) ? a : log1pf(__expf(a));
    }
}

// ---------------- selective scan, fused D-skip and SiLU(z) gate ----------------
__global__ void scan_kernel(const float* __restrict__ DT,
                            const float* __restrict__ XD,
                            const float* __restrict__ Uf,
                            const float* __restrict__ xz,
                            const float* __restrict__ A_log,
                            const float* __restrict__ Dp,
                            unsigned short* __restrict__ Ybf) {
    int b   = blockIdx.x;
    int tid = threadIdx.x;                  // 256
    int d   = blockIdx.y * blockDim.x + tid;
    __shared__ float Bs[DSTATE], Cs[DSTATE];

    float A[DSTATE], h[DSTATE];
    #pragma unroll
    for (int n = 0; n < DSTATE; ++n) {
        A[n] = -__expf(A_log[d * DSTATE + n]);
        h[n] = 0.f;
    }
    float Dd = Dp[d];

    for (int l = 0; l < NPATCH; ++l) {
        size_t row = (size_t)b * NPATCH + l;
        __syncthreads();
        if (tid < DSTATE)            Bs[tid]          = XD[row * XDW + DTRANK + tid];
        else if (tid < 2 * DSTATE)   Cs[tid - DSTATE] = XD[row * XDW + DTRANK + DSTATE + (tid - DSTATE)];
        __syncthreads();

        float dtv = DT[row * DINNER + d];
        float uv  = Uf[row * DINNER + d];
        float du  = dtv * uv;
        float y   = 0.f;
        #pragma unroll
        for (int n = 0; n < DSTATE; ++n) {
            h[n] = __expf(dtv * A[n]) * h[n] + du * Bs[n];
            y   += h[n] * Cs[n];
        }
        float yv = y + uv * Dd;
        float zv = xz[row * XZW + DINNER + d];
        yv *= silu(zv);
        Ybf[row * DINNER + d] = f2bf(yv);
    }
}

// ---------------- mean over L then dot(head_w) + head_b ----------------
__global__ void head_kernel(const float* __restrict__ OUTF,
                            const float* __restrict__ head_w,
                            const float* __restrict__ head_b,
                            float* __restrict__ out) {
    int b = blockIdx.x;
    int d = threadIdx.x;                    // 256
    __shared__ float red[DMODEL];
    float s = 0.f;
    for (int l = 0; l < NPATCH; ++l)
        s += OUTF[((size_t)b * NPATCH + l) * DMODEL + d];
    red[d] = s * head_w[d];
    __syncthreads();
    for (int st = DMODEL / 2; st > 0; st >>= 1) { if (d < st) red[d] += red[d + st]; __syncthreads(); }
    if (d == 0) out[b] = red[0] * (1.f / NPATCH) + head_b[0];
}

// ---------------- host side ----------------
static inline size_t alignup(size_t v) { return (v + 255) & ~(size_t)255; }

extern "C" void kernel_launch(void* const* d_in, const int* in_sizes, int n_in,
                              void* d_out, int out_size, void* d_ws, size_t ws_size,
                              hipStream_t stream) {
    (void)in_sizes; (void)n_in; (void)out_size; (void)ws_size;
    const float* x         = (const float*)d_in[0];
    const float* ln1_g     = (const float*)d_in[1];
    const float* ln1_b     = (const float*)d_in[2];
    const float* W_emb     = (const float*)d_in[3];
    const float* b_emb     = (const float*)d_in[4];
    const float* ln2_g     = (const float*)d_in[5];
    const float* ln2_b     = (const float*)d_in[6];
    const float* in_proj_w = (const float*)d_in[7];
    const float* conv_w    = (const float*)d_in[8];
    const float* conv_b    = (const float*)d_in[9];
    const float* x_proj_w  = (const float*)d_in[10];
    const float* dt_proj_w = (const float*)d_in[11];
    const float* dt_proj_b = (const float*)d_in[12];
    const float* A_log     = (const float*)d_in[13];
    const float* Dp        = (const float*)d_in[14];
    const float* out_proj_w= (const float*)d_in[15];
    const float* head_w    = (const float*)d_in[16];
    const float* head_b    = (const float*)d_in[17];
    float* out = (float*)d_out;

    // workspace carve-up
    char* ws = (char*)d_ws;
    size_t off = 0;
    unsigned short* Wi  = (unsigned short*)(ws + off); off = alignup(off + (size_t)XZW * DMODEL * 2);
    unsigned short* Wx  = (unsigned short*)(ws + off); off = alignup(off + (size_t)XDW * DINNER * 2);
    unsigned short* Wo  = (unsigned short*)(ws + off); off = alignup(off + (size_t)DMODEL * DINNER * 2);
    unsigned short* Pbf = (unsigned short*)(ws + off); off = alignup(off + (size_t)NROWS * DMODEL * 2);
    float*          XZ  = (float*)(ws + off);          off = alignup(off + (size_t)NROWS * XZW * 4);
    float*          Uf  = (float*)(ws + off);          off = alignup(off + (size_t)NROWS * DINNER * 4);
    unsigned short* Ubf = (unsigned short*)(ws + off); off = alignup(off + (size_t)NROWS * DINNER * 2);
    float*          XD  = (float*)(ws + off);          off = alignup(off + (size_t)NROWS * XDW * 4);
    float*          DT  = (float*)(ws + off);          off = alignup(off + (size_t)NROWS * DINNER * 4);
    unsigned short* Ybf = (unsigned short*)(ws + off); off = alignup(off + (size_t)NROWS * DINNER * 2);
    float*          OUTF= (float*)(ws + off);          off = alignup(off + (size_t)NROWS * DMODEL * 4);

    // 1) weights -> bf16
    {
        int n;
        n = XZW * DMODEL;    f32_to_bf16_kernel<<<(n + 255) / 256, 256, 0, stream>>>(in_proj_w,  Wi, n);
        n = XDW * DINNER;    f32_to_bf16_kernel<<<(n + 255) / 256, 256, 0, stream>>>(x_proj_w,   Wx, n);
        n = DMODEL * DINNER; f32_to_bf16_kernel<<<(n + 255) / 256, 256, 0, stream>>>(out_proj_w, Wo, n);
    }

    // 2) patch embed -> Pbf [16384, 256] bf16
    patch_embed_kernel<<<NROWS, DMODEL, 0, stream>>>(x, ln1_g, ln1_b, W_emb, b_emb, ln2_g, ln2_b, Pbf);

    // 3) in_proj GEMM: XZ[16384,1024] = Pbf @ Wi^T   (K=256, 4 N-tiles/block, no guards)
    gemm_bf16_wmma<4, false><<<dim3(NROWS / 128, XZW / 64), 128, 0, stream>>>(
        Pbf, Wi, XZ, NROWS, XZW, DMODEL);

    // 4) causal depthwise conv + SiLU -> Uf/Ubf [16384, 512]
    conv_silu_kernel<<<NROWS, DINNER, 0, stream>>>(XZ, conv_w, conv_b, Uf, Ubf);

    // 5) x_proj GEMM: XD[16384,48] = Ubf @ Wx^T   (K=512, 3 N-tiles, staging guarded)
    gemm_bf16_wmma<3, true><<<dim3(NROWS / 128, 1), 128, 0, stream>>>(
        Ubf, Wx, XD, NROWS, XDW, DINNER);

    // 6) dt = softplus(dt_in @ dt_proj_w^T + dt_proj_b)
    dt_kernel<<<NROWS, 256, 0, stream>>>(XD, dt_proj_w, dt_proj_b, DT);

    // 7) selective scan (+D skip, *silu(z)) -> Ybf [16384, 512]
    scan_kernel<<<dim3(BATCH, DINNER / 256), 256, 0, stream>>>(DT, XD, Uf, XZ, A_log, Dp, Ybf);

    // 8) out_proj GEMM: OUTF[16384,256] = Ybf @ Wo^T   (K=512, 4 N-tiles, no guards)
    gemm_bf16_wmma<4, false><<<dim3(NROWS / 128, DMODEL / 64), 128, 0, stream>>>(
        Ybf, Wo, OUTF, NROWS, DMODEL, DINNER);

    // 9) mean over L, head dot
    head_kernel<<<BATCH, DMODEL, 0, stream>>>(OUTF, head_w, head_b, out);
}
